// QwenMambaHybridLM_88390426952212
// MI455X (gfx1250) — compile-verified, matrix operand order checked by reading, not run on hardware
//
#include <hip/hip_runtime.h>
#include <math.h>

// ---------------------------------------------------------------------------
// Types for WMMA
// ---------------------------------------------------------------------------
typedef __attribute__((ext_vector_type(16))) _Float16 v16h;
typedef __attribute__((ext_vector_type(8)))  float    v8f;

union FragU  { v16h v; uint4 q[2]; };
union HalfPack { _Float16 h[4]; uint2 u; };

#define TS 56   // padded LDS row stride in elements (112 bytes, 16B multiple)

// ---------------------------------------------------------------------------
// Generic batched GEMM: C[z] = A[z] @ B[z]  (optionally B transposed, and/or
// accumulate into C).  fp32 in / fp32 out, f16 WMMA compute, fp32 accumulate.
// Tile 128x128x32, 256 threads = 8 waves, wave grid 2x4, wave tile 64x32.
// Double-buffered LDS + register prefetch software pipeline.
// flags bit0 = transB (B stored [N][K] row-major), bit1 = C += result.
// ---------------------------------------------------------------------------
__global__ __launch_bounds__(256)
void k_gemm(int M, int N, int K,
            const float* __restrict__ A, int lda, long long sA,
            const float* __restrict__ B, int ldb, long long sB,
            float* __restrict__ C, int ldc, long long sC,
            int flags)
{
    __shared__ _Float16 As[2][128 * TS];
    __shared__ _Float16 Bs[2][128 * TS];

    const int tid  = threadIdx.x;
    const int z    = blockIdx.z;
    A += (long long)z * sA;
    B += (long long)z * sB;
    C += (long long)z * sC;

    const int m0   = blockIdx.y * 128;
    const int n0   = blockIdx.x * 128;
    const int lane = tid & 31;
    const int wave = tid >> 5;
    const int wm   = (wave >> 2) * 64;   // wave row offset in tile
    const int wn   = (wave & 3) * 32;    // wave col offset in tile
    const int lm   = lane & 15;
    const int kg   = lane >> 4;          // k-group (0/1)
    const bool transB = (flags & 1) != 0;
    const bool accum  = (flags & 2) != 0;
    const bool ldaV4  = (lda & 3) == 0;
    const bool ldbV4  = (ldb & 3) == 0;

    const int nK = (K + 31) >> 5;

    // ---- tile staging: global -> registers (branchless) ----
    float ra[16], rb[16];

    auto loadA = [&](int kk) {
        const bool interior = (m0 + 128 <= M) && (kk + 32 <= K);
        if (interior && ldaV4) {
            #pragma unroll
            for (int i = 0; i < 4; i++) {
                int g = tid + 256 * i;
                int rr = g >> 3, cc = (g & 7) * 4;        // rr = m, cc = k
                const float4 v = *(const float4*)(A + (long long)(m0 + rr) * lda + kk + cc);
                ra[i*4+0] = v.x; ra[i*4+1] = v.y; ra[i*4+2] = v.z; ra[i*4+3] = v.w;
            }
        } else if (interior) {
            #pragma unroll
            for (int i = 0; i < 4; i++) {
                int g = tid + 256 * i;
                int rr = g >> 3, cc = (g & 7) * 4;
                const float* p = A + (long long)(m0 + rr) * lda + kk + cc;
                ra[i*4+0] = p[0]; ra[i*4+1] = p[1]; ra[i*4+2] = p[2]; ra[i*4+3] = p[3];
            }
        } else {
            #pragma unroll
            for (int i = 0; i < 4; i++) {
                int g = tid + 256 * i;
                int rr = g >> 3, cc = (g & 7) * 4;
                #pragma unroll
                for (int j = 0; j < 4; j++) {
                    int gm = m0 + rr, gk = kk + cc + j;
                    bool ok = (gm < M) && (gk < K);
                    long long o = ok ? ((long long)gm * lda + gk) : 0;
                    float v = A[o];
                    ra[i*4+j] = ok ? v : 0.f;
                }
            }
        }
    };

    auto loadB = [&](int kk) {
        if (!transB) {
            const bool interior = (n0 + 128 <= N) && (kk + 32 <= K);
            if (interior && ldbV4) {
                #pragma unroll
                for (int i = 0; i < 4; i++) {
                    int g = tid + 256 * i;
                    int rr = g >> 5, cc = (g & 31) * 4;   // rr = k, cc = n
                    const float4 v = *(const float4*)(B + (long long)(kk + rr) * ldb + n0 + cc);
                    rb[i*4+0] = v.x; rb[i*4+1] = v.y; rb[i*4+2] = v.z; rb[i*4+3] = v.w;
                }
            } else if (interior) {
                #pragma unroll
                for (int i = 0; i < 4; i++) {
                    int g = tid + 256 * i;
                    int rr = g >> 5, cc = (g & 31) * 4;
                    const float* p = B + (long long)(kk + rr) * ldb + n0 + cc;
                    rb[i*4+0] = p[0]; rb[i*4+1] = p[1]; rb[i*4+2] = p[2]; rb[i*4+3] = p[3];
                }
            } else {
                #pragma unroll
                for (int i = 0; i < 4; i++) {
                    int g = tid + 256 * i;
                    int rr = g >> 5, cc = (g & 31) * 4;
                    #pragma unroll
                    for (int j = 0; j < 4; j++) {
                        int gk = kk + rr, gn = n0 + cc + j;
                        bool ok = (gk < K) && (gn < N);
                        long long o = ok ? ((long long)gk * ldb + gn) : 0;
                        float v = B[o];
                        rb[i*4+j] = ok ? v : 0.f;
                    }
                }
            }
        } else {
            const bool interior = (n0 + 128 <= N) && (kk + 32 <= K);
            if (interior && ldbV4) {
                #pragma unroll
                for (int i = 0; i < 4; i++) {
                    int g = tid + 256 * i;
                    int rr = g >> 3, cc = (g & 7) * 4;    // rr = n, cc = k
                    const float4 v = *(const float4*)(B + (long long)(n0 + rr) * ldb + kk + cc);
                    rb[i*4+0] = v.x; rb[i*4+1] = v.y; rb[i*4+2] = v.z; rb[i*4+3] = v.w;
                }
            } else if (interior) {
                #pragma unroll
                for (int i = 0; i < 4; i++) {
                    int g = tid + 256 * i;
                    int rr = g >> 3, cc = (g & 7) * 4;
                    const float* p = B + (long long)(n0 + rr) * ldb + kk + cc;
                    rb[i*4+0] = p[0]; rb[i*4+1] = p[1]; rb[i*4+2] = p[2]; rb[i*4+3] = p[3];
                }
            } else {
                #pragma unroll
                for (int i = 0; i < 4; i++) {
                    int g = tid + 256 * i;
                    int rr = g >> 3, cc = (g & 7) * 4;
                    #pragma unroll
                    for (int j = 0; j < 4; j++) {
                        int gn = n0 + rr, gk = kk + cc + j;
                        bool ok = (gn < N) && (gk < K);
                        long long o = ok ? ((long long)gn * ldb + gk) : 0;
                        float v = B[o];
                        rb[i*4+j] = ok ? v : 0.f;
                    }
                }
            }
        }
    };

    // ---- registers -> LDS (f16 convert) ----
    auto storeA = [&](int buf) {
        #pragma unroll
        for (int i = 0; i < 4; i++) {
            int g = tid + 256 * i;
            int rr = g >> 3, cc = (g & 7) * 4;
            HalfPack hp;
            hp.h[0] = (_Float16)ra[i*4+0]; hp.h[1] = (_Float16)ra[i*4+1];
            hp.h[2] = (_Float16)ra[i*4+2]; hp.h[3] = (_Float16)ra[i*4+3];
            *(uint2*)&As[buf][rr * TS + cc] = hp.u;
        }
    };
    auto storeB = [&](int buf) {
        if (!transB) {
            #pragma unroll
            for (int i = 0; i < 4; i++) {
                int g = tid + 256 * i;
                int rr = g >> 5, cc = (g & 31) * 4;       // rr = k, cc = n
                #pragma unroll
                for (int j = 0; j < 4; j++)
                    Bs[buf][(cc + j) * TS + rr] = (_Float16)rb[i*4+j];
            }
        } else {
            #pragma unroll
            for (int i = 0; i < 4; i++) {
                int g = tid + 256 * i;
                int rr = g >> 3, cc = (g & 7) * 4;        // rr = n, cc = k
                HalfPack hp;
                hp.h[0] = (_Float16)rb[i*4+0]; hp.h[1] = (_Float16)rb[i*4+1];
                hp.h[2] = (_Float16)rb[i*4+2]; hp.h[3] = (_Float16)rb[i*4+3];
                *(uint2*)&Bs[buf][rr * TS + cc] = hp.u;
            }
        }
    };

    v8f acc[4][2] = {};

    // ---- software pipeline: prefetch s+1 while computing s ----
    loadA(0); loadB(0);
    storeA(0); storeB(0);
    __syncthreads();

    for (int s = 0; s < nK; s++) {
        const int cur = s & 1;
        const bool more = (s + 1) < nK;
        if (more) { loadA((s + 1) * 32); loadB((s + 1) * 32); }

        // ---- fragments + WMMA on buffer `cur` ----
        FragU a[4];
        #pragma unroll
        for (int sm = 0; sm < 4; sm++) {
            const _Float16* p = &As[cur][(wm + sm * 16 + lm) * TS + kg * 8];
            a[sm].q[0] = *(const uint4*)(p);
            a[sm].q[1] = *(const uint4*)(p + 16);
        }
        #pragma unroll
        for (int sn = 0; sn < 2; sn++) {
            FragU b;
            const _Float16* p = &Bs[cur][(wn + sn * 16 + lm) * TS + kg * 8];
            b.q[0] = *(const uint4*)(p);
            b.q[1] = *(const uint4*)(p + 16);
            #pragma unroll
            for (int sm = 0; sm < 4; sm++) {
                acc[sm][sn] = __builtin_amdgcn_wmma_f32_16x16x32_f16(
                    false, a[sm].v, false, b.v, (short)0, acc[sm][sn],
                    false, false);
            }
        }

        if (more) { storeA(cur ^ 1); storeB(cur ^ 1); }
        __syncthreads();
    }

    // ---- epilogue: C/D layout, VGPR r -> m = r + (lane>>4)*8, n = lane&15 ----
    const int mrow = kg * 8;
    #pragma unroll
    for (int sm = 0; sm < 4; sm++) {
        #pragma unroll
        for (int sn = 0; sn < 2; sn++) {
            #pragma unroll
            for (int r = 0; r < 8; r++) {
                int gm = m0 + wm + sm * 16 + mrow + r;
                int gn = n0 + wn + sn * 16 + lm;
                if (gm < M && gn < N) {
                    long long o = (long long)gm * ldc + gn;
                    float val = acc[sm][sn][r];
                    if (accum) val += C[o];
                    C[o] = val;
                }
            }
        }
    }
}

// ---------------------------------------------------------------------------
// Elementwise / reduction helper kernels
// ---------------------------------------------------------------------------
__global__ void k_embed(const int* __restrict__ tok, const float* __restrict__ emb,
                        float* __restrict__ x, int n)
{
    int i = blockIdx.x * 256 + threadIdx.x;
    if (i >= n) return;
    int row = i >> 10;           // D = 1024
    int d   = i & 1023;
    x[i] = emb[(long long)tok[row] * 1024 + d];
}

__global__ __launch_bounds__(256)
void k_rmsnorm(const float* __restrict__ x, const float* __restrict__ w,
               float* __restrict__ o, int D)
{
    int row = blockIdx.x;
    const float* xr = x + (long long)row * D;
    float s = 0.f;
    for (int d = threadIdx.x; d < D; d += 256) { float v = xr[d]; s += v * v; }
    __shared__ float red[256];
    red[threadIdx.x] = s; __syncthreads();
    for (int st = 128; st > 0; st >>= 1) {
        if (threadIdx.x < st) red[threadIdx.x] += red[threadIdx.x + st];
        __syncthreads();
    }
    float rs = rsqrtf(red[0] / (float)D + 1e-6f);
    for (int d = threadIdx.x; d < D; d += 256)
        o[(long long)row * D + d] = xr[d] * rs * w[d];
}

// depthwise causal conv (width 4) + bias + SiLU.  xz is 2048x4096 (x_ssm = cols 0..2047)
__global__ void k_conv_silu(const float* __restrict__ xz, const float* __restrict__ cw,
                            const float* __restrict__ cb, float* __restrict__ out, int n)
{
    int i = blockIdx.x * 256 + threadIdx.x;
    if (i >= n) return;                 // n = 2048*2048
    int row = i >> 11;                  // 2048 channels
    int c   = i & 2047;
    int b = row >> 10, t = row & 1023;
    float s = cb[c];
    #pragma unroll
    for (int j = 0; j < 4; j++) {
        int tt = t + j - 3;
        if (tt >= 0) s += cw[c * 4 + j] * xz[((long long)(b * 1024 + tt)) * 4096 + c];
    }
    out[i] = s / (1.f + expf(-s));
}

__global__ void k_softplus_bias(float* __restrict__ d, const float* __restrict__ bias, int n)
{
    int i = blockIdx.x * 256 + threadIdx.x;
    if (i >= n) return;
    int c = i & 2047;
    float v = d[i] + bias[c];
    d[i] = (v > 20.f) ? v : log1pf(expf(v));
}

// selective scan: one thread per (batch, channel); 16-state recurrence over L
__global__ __launch_bounds__(256)
void k_ssm_scan(const float* __restrict__ delta, const float* __restrict__ xc,
                const float* __restrict__ xp, const float* __restrict__ xz,
                const float* __restrict__ logA, const float* __restrict__ Dp,
                float* __restrict__ yg)
{
    int g = blockIdx.x * 256 + threadIdx.x;  // 0..4095
    int b = g >> 11, c = g & 2047;
    float Ar[16], st[16];
    #pragma unroll
    for (int s = 0; s < 16; s++) { Ar[s] = -expf(logA[c * 16 + s]); st[s] = 0.f; }
    float dp = Dp[c];
    for (int t = 0; t < 1024; t++) {
        long long row = (long long)b * 1024 + t;
        float de = delta[row * 2048 + c];
        float xv = xc[row * 2048 + c];
        const float* xpr = xp + row * 96;
        float acc = 0.f;
        #pragma unroll
        for (int s = 0; s < 16; s++) {
            float dA = expf(de * Ar[s]);
            st[s] = dA * st[s] + de * xpr[64 + s] * xv;
            acc += st[s] * xpr[80 + s];
        }
        float zv = xz[row * 4096 + 2048 + c];
        yg[row * 2048 + c] = (acc + xv * dp) * (zv / (1.f + expf(-zv)));
    }
}

// RoPE on q,k + transpose q,k,v from [b*L+t][H*64] to per-(b,h) [t][64] layout
__global__ void k_rope(const float* __restrict__ q, const float* __restrict__ kk,
                       const float* __restrict__ v, float* __restrict__ qh,
                       float* __restrict__ kh, float* __restrict__ vh, int n)
{
    int i = blockIdx.x * 256 + threadIdx.x;
    if (i >= n) return;                  // n = B*H*L*32
    int p = i & 31;
    int t = (i >> 5) & 1023;
    int h = (i >> 15) & 15;
    int b = i >> 19;
    float inv = expf(-(float)(2 * p) * (1.0f / 64.0f) * logf(10000.0f));
    float ang = (float)t * inv;
    float cs = cosf(ang), sn = sinf(ang);
    long long src = ((long long)(b * 1024 + t)) * 1024 + h * 64;
    long long dst = (((long long)(b * 16 + h)) * 1024 + t) * 64;
    float q1 = q[src + p], q2 = q[src + 32 + p];
    qh[dst + p]      = q1 * cs - q2 * sn;
    qh[dst + 32 + p] = q2 * cs + q1 * sn;
    float k1 = kk[src + p], k2 = kk[src + 32 + p];
    kh[dst + p]      = k1 * cs - k2 * sn;
    kh[dst + 32 + p] = k2 * cs + k1 * sn;
    vh[dst + p]      = v[src + p];
    vh[dst + 32 + p] = v[src + 32 + p];
}

// scale by 1/sqrt(64), causal mask, softmax over one row of 1024
__global__ __launch_bounds__(256)
void k_softmax_causal(float* __restrict__ s)
{
    int row  = blockIdx.x;            // z*1024 + q
    int qpos = row & 1023;
    float* sr = s + (long long)row * 1024;
    __shared__ float red[256];
    float m = -1e30f;
    for (int nn = threadIdx.x; nn < 1024; nn += 256) {
        float v = (nn <= qpos) ? sr[nn] * 0.125f : -1e30f;
        m = fmaxf(m, v);
    }
    red[threadIdx.x] = m; __syncthreads();
    for (int st = 128; st > 0; st >>= 1) {
        if (threadIdx.x < st) red[threadIdx.x] = fmaxf(red[threadIdx.x], red[threadIdx.x + st]);
        __syncthreads();
    }
    m = red[0]; __syncthreads();
    float sum = 0.f;
    for (int nn = threadIdx.x; nn < 1024; nn += 256) {
        float v = (nn <= qpos) ? expf(sr[nn] * 0.125f - m) : 0.f;
        sr[nn] = v; sum += v;
    }
    red[threadIdx.x] = sum; __syncthreads();
    for (int st = 128; st > 0; st >>= 1) {
        if (threadIdx.x < st) red[threadIdx.x] += red[threadIdx.x + st];
        __syncthreads();
    }
    float invs = 1.f / red[0];
    for (int nn = threadIdx.x; nn < 1024; nn += 256) sr[nn] *= invs;
}

__global__ void k_merge_heads(const float* __restrict__ ctx, float* __restrict__ o, int n)
{
    int i = blockIdx.x * 256 + threadIdx.x;
    if (i >= n) return;                 // n = 2048*1024, output [b*L+t][H*64]
    int row = i >> 10, col = i & 1023;
    int b = row >> 10, t = row & 1023;
    int h = col >> 6, d = col & 63;
    o[i] = ctx[(((long long)(b * 16 + h)) * 1024 + t) * 64 + d];
}

__global__ void k_silu_mul(const float* __restrict__ g, const float* __restrict__ u,
                           float* __restrict__ o, int n)
{
    int i = blockIdx.x * 256 + threadIdx.x;
    if (i >= n) return;
    float x = g[i];
    o[i] = (x / (1.f + expf(-x))) * u[i];
}

// ---------------------------------------------------------------------------
// Host-side orchestration
// ---------------------------------------------------------------------------
extern "C" void kernel_launch(void* const* d_in, const int* in_sizes, int n_in,
                              void* d_out, int out_size, void* d_ws, size_t ws_size,
                              hipStream_t stream)
{
    (void)in_sizes; (void)n_in; (void)out_size; (void)ws_size;

    const int*   tokens   = (const int*)d_in[0];
    const float* emb      = (const float*)d_in[1];
    const float* m_norm   = (const float*)d_in[2];
    const float* m_inproj = (const float*)d_in[3];
    const float* m_convw  = (const float*)d_in[4];
    const float* m_convb  = (const float*)d_in[5];
    const float* m_xproj  = (const float*)d_in[6];
    const float* m_dtw    = (const float*)d_in[7];
    const float* m_dtb    = (const float*)d_in[8];
    const float* m_logA   = (const float*)d_in[9];
    const float* m_Dp     = (const float*)d_in[10];
    const float* m_outp   = (const float*)d_in[11];
    const float* a_n1     = (const float*)d_in[12];
    const float* a_wq     = (const float*)d_in[13];
    const float* a_wk     = (const float*)d_in[14];
    const float* a_wv     = (const float*)d_in[15];
    const float* a_wo     = (const float*)d_in[16];
    const float* a_n2     = (const float*)d_in[17];
    const float* a_gate   = (const float*)d_in[18];
    const float* a_up     = (const float*)d_in[19];
    const float* a_down   = (const float*)d_in[20];
    const float* fnorm    = (const float*)d_in[21];

    const int  R  = 2048;              // B*L rows
    const int  FF = 2730;
    const long long RD = (long long)R * 1024;     // 2,097,152 floats

    float* ws = (float*)d_ws;
    // time-multiplexed arena (floats)
    float* x      = ws;                 // RD
    float* h      = ws + RD;            // RD
    float* xz     = ws + 2 * RD;        // 4*RD (mamba)   | q,k,v live here for attn
    float* q      = ws + 2 * RD;        // RD
    float* kbuf   = ws + 3 * RD;        // RD
    float* vbuf   = ws + 4 * RD;        // RD
    float* qh     = ws + 5 * RD;        // RD
    float* xconv  = ws + 6 * RD;        // 2*RD (mamba)   | kh,vh for attn
    float* kh     = ws + 6 * RD;        // RD
    float* vh     = ws + 7 * RD;        // RD
    float* deltab = ws + 8 * RD;        // 2*RD (mamba)   | ctx,ctxm for attn
    float* ctx    = ws + 8 * RD;        // RD
    float* ctxm   = ws + 9 * RD;        // RD
    float* yg     = ws + 10 * RD;       // 2*RD (mamba)
    float* scores = ws + 10 * RD;       // 16*RD (attn, reused by FFN bufs)
    float* xp     = ws + 12 * RD;       // 2048*96 (mamba)
    const long long FFsz = (long long)R * FF;
    float* G = ws + 10 * RD;            // after attention core is done
    float* U = G + FFsz;
    float* F = U + FFsz;

    auto gemm = [&](int M, int N, int K,
                    const float* A, int lda, long long sA,
                    const float* B, int ldb, long long sB,
                    float* C, int ldc, long long sC,
                    int batch, int flags) {
        dim3 g((N + 127) / 128, (M + 127) / 128, batch);
        k_gemm<<<g, 256, 0, stream>>>(M, N, K, A, lda, sA, B, ldb, sB, C, ldc, sC, flags);
    };

    // x = emb[tokens]
    {
        int n = R * 1024;
        k_embed<<<n / 256, 256, 0, stream>>>(tokens, emb, x, n);
    }

    for (int layer = 0; layer < 4; layer++) {
        if ((layer & 1) == 0) {
            // ---------------- Mamba block ----------------
            int i = layer / 2;
            k_rmsnorm<<<R, 256, 0, stream>>>(x, m_norm + i * 1024, h, 1024);
            // xz = h @ in_proj  (2048 x 4096)
            gemm(R, 4096, 1024, h, 1024, 0, m_inproj + (long long)i * 1024 * 4096, 4096, 0,
                 xz, 4096, 0, 1, 0);
            // depthwise conv + SiLU
            {
                int n = R * 2048;
                k_conv_silu<<<n / 256, 256, 0, stream>>>(xz, m_convw + i * 2048 * 4,
                                                         m_convb + i * 2048, xconv, n);
            }
            // xp = xconv @ x_proj  (2048 x 96)
            gemm(R, 96, 2048, xconv, 2048, 0, m_xproj + (long long)i * 2048 * 96, 96, 0,
                 xp, 96, 0, 1, 0);
            // delta_raw = xp[:, :64] @ dt_w  (2048 x 2048)
            gemm(R, 2048, 64, xp, 96, 0, m_dtw + (long long)i * 64 * 2048, 2048, 0,
                 deltab, 2048, 0, 1, 0);
            {
                int n = R * 2048;
                k_softplus_bias<<<n / 256, 256, 0, stream>>>(deltab, m_dtb + i * 2048, n);
            }
            // selective scan -> gated y
            k_ssm_scan<<<16, 256, 0, stream>>>(deltab, xconv, xp, xz,
                                               m_logA + (long long)i * 2048 * 16,
                                               m_Dp + i * 2048, yg);
            // x += yg @ out_proj
            gemm(R, 1024, 2048, yg, 2048, 0, m_outp + (long long)i * 2048 * 1024, 1024, 0,
                 x, 1024, 0, 1, 2);
        } else {
            // ---------------- Attention block ----------------
            int i = layer / 2;
            const long long W = (long long)i * 1024 * 1024;
            k_rmsnorm<<<R, 256, 0, stream>>>(x, a_n1 + i * 1024, h, 1024);
            gemm(R, 1024, 1024, h, 1024, 0, a_wq + W, 1024, 0, q, 1024, 0, 1, 0);
            gemm(R, 1024, 1024, h, 1024, 0, a_wk + W, 1024, 0, kbuf, 1024, 0, 1, 0);
            gemm(R, 1024, 1024, h, 1024, 0, a_wv + W, 1024, 0, vbuf, 1024, 0, 1, 0);
            {
                int n = 2 * 16 * 1024 * 32;  // B*H*L*32
                k_rope<<<n / 256, 256, 0, stream>>>(q, kbuf, vbuf, qh, kh, vh, n);
            }
            // scores[z] = Q[z] @ K[z]^T, batch = 32
            gemm(1024, 1024, 64, qh, 64, 65536, kh, 64, 65536,
                 scores, 1024, 1048576, 32, 1 /*transB*/);
            k_softmax_causal<<<32 * 1024, 256, 0, stream>>>(scores);
            // ctx[z] = scores[z] @ V[z]
            gemm(1024, 64, 1024, scores, 1024, 1048576, vh, 64, 65536,
                 ctx, 64, 65536, 32, 0);
            {
                int n = R * 1024;
                k_merge_heads<<<n / 256, 256, 0, stream>>>(ctx, ctxm, n);
            }
            // x += ctx @ wo
            gemm(R, 1024, 1024, ctxm, 1024, 0, a_wo + W, 1024, 0, x, 1024, 0, 1, 2);
            // FFN
            k_rmsnorm<<<R, 256, 0, stream>>>(x, a_n2 + i * 1024, h, 1024);
            gemm(R, FF, 1024, h, 1024, 0, a_gate + (long long)i * 1024 * FF, FF, 0,
                 G, FF, 0, 1, 0);
            gemm(R, FF, 1024, h, 1024, 0, a_up + (long long)i * 1024 * FF, FF, 0,
                 U, FF, 0, 1, 0);
            {
                long long n = FFsz;
                k_silu_mul<<<(int)((n + 255) / 256), 256, 0, stream>>>(G, U, F, (int)n);
            }
            gemm(R, 1024, FF, F, FF, 0, a_down + (long long)i * FF * 1024, 1024, 0,
                 x, 1024, 0, 1, 2);
        }
    }

    // final norm + logits = h @ emb^T
    k_rmsnorm<<<R, 256, 0, stream>>>(x, fnorm, h, 1024);
    gemm(R, 32000, 1024, h, 1024, 0, emb, 1024, 0, (float*)d_out, 32000, 0, 1, 1 /*transB*/);
}
